// ViS4mer_52209622450649
// MI455X (gfx1250) — compile-verified, matrix operand order checked by reading, not run on hardware
//
#include <hip/hip_runtime.h>
#include <hip/hip_bf16.h>
#include <math.h>

// ---------------------------------------------------------------------------
// Types for CDNA5 WMMA (gfx1250, wave32)
// ---------------------------------------------------------------------------
typedef __attribute__((ext_vector_type(16))) __bf16 v16bf;
typedef __attribute__((ext_vector_type(8)))  float  v8f;

// K-pair base for 16-bit A/B fragments (ISA 7.12.2: VGPR j holds K={kb,kb+1},
// lanes 0-15 cover K 0..7 / 16..23, lanes 16-31 cover K 8..15 / 24..31)
__device__ __forceinline__ int wmma_kbase(int j, int hi) {
  return (j < 4) ? (hi * 8 + 2 * j) : (16 + hi * 8 + 2 * (j - 4));
}

__device__ __forceinline__ float gelu_exact(float x) {
  return 0.5f * x * (1.0f + erff(x * 0.70710678118654752f));
}
__device__ __forceinline__ float sigmoidf(float x) {
  return 1.0f / (1.0f + expf(-x));
}

// ---------------------------------------------------------------------------
// 1) Transpose x1 (B,L,H) -> x (B,H,L), tiled 32x32 through LDS
// ---------------------------------------------------------------------------
__global__ void transpose_blh_to_bhl(const float* __restrict__ in,
                                     float* __restrict__ out, int H, int L) {
  __shared__ float tile[32][33];
  int b  = blockIdx.z;
  int h0 = blockIdx.x * 32;
  int l0 = blockIdx.y * 32;
  int lx = threadIdx.x;  // 0..31
  int ty = threadIdx.y;  // 0..7
  for (int i = ty; i < 32; i += 8)
    tile[i][lx] = in[((size_t)b * L + (l0 + i)) * H + h0 + lx];  // [l][h]
  __syncthreads();
  for (int i = ty; i < 32; i += 8)
    out[((size_t)b * H + (h0 + i)) * L + l0 + lx] = tile[lx][i];
}

// ---------------------------------------------------------------------------
// 2) LayerNorm over H for each (b,l); data in (B,H,L). 256 threads = 32 l x 8 h
// ---------------------------------------------------------------------------
__global__ void __launch_bounds__(256)
layernorm_bhl(const float* __restrict__ x, const float* __restrict__ g,
              const float* __restrict__ beta, float* __restrict__ out,
              int H, int L) {
  __shared__ float s_sum[8][32], s_sq[8][32], s_mu[32], s_rs[32];
  int b    = blockIdx.y;
  int lane = threadIdx.x & 31;
  int ty   = threadIdx.x >> 5;
  int l    = blockIdx.x * 32 + lane;
  const float* xb = x + (size_t)b * H * L;
  float sum = 0.f, sq = 0.f;
  for (int h = ty; h < H; h += 8) {
    float v = xb[(size_t)h * L + l];
    sum += v; sq += v * v;
  }
  s_sum[ty][lane] = sum; s_sq[ty][lane] = sq;
  __syncthreads();
  if (ty == 0) {
    float S = 0.f, Q = 0.f;
    #pragma unroll
    for (int i = 0; i < 8; i++) { S += s_sum[i][lane]; Q += s_sq[i][lane]; }
    float mu  = S / (float)H;
    float var = Q / (float)H - mu * mu;
    s_mu[lane] = mu;
    s_rs[lane] = rsqrtf(var + 1e-5f);
  }
  __syncthreads();
  float mu = s_mu[lane], rs = s_rs[lane];
  float* ob = out + (size_t)b * H * L;
  for (int h = ty; h < H; h += 8) {
    float v = xb[(size_t)h * L + l];
    ob[(size_t)h * L + l] = (v - mu) * rs * g[h] + beta[h];
  }
}

// ---------------------------------------------------------------------------
// 3) S4D as linear recurrence, in-place on (B,H,L). One wave per (b,h) row,
//    32 lanes x 2 complex states = N=64. Output = gelu(y_conv + D*u).
// ---------------------------------------------------------------------------
__global__ void __launch_bounds__(256)
s4_scan(float* __restrict__ xn, const float* __restrict__ log_dt,
        const float* __restrict__ C_re, const float* __restrict__ C_im,
        const float* __restrict__ Dv, int H, int L) {
  int lane = threadIdx.x & 31;
  int wid  = blockIdx.x * (blockDim.x >> 5) + (threadIdx.x >> 5);
  int b = wid / H, h = wid % H;
  const float PI = 3.14159265358979323846f;
  float dt = expf(log_dt[h]);
  float w_re[2], w_im[2], cd_re[2], cd_im[2], s_re[2], s_im[2];
  #pragma unroll
  for (int q = 0; q < 2; q++) {
    int   n    = lane + 32 * q;
    float a_re = -0.5f, a_im = PI * (float)n;
    float dre = dt * a_re, dim = dt * a_im;
    float e  = expf(dre);
    float wr = e * cosf(dim), wi = e * sinf(dim);
    w_re[q] = wr; w_im[q] = wi;
    float em_re = wr - 1.0f, em_im = wi;              // exp(dtA) - 1
    float den   = a_re * a_re + a_im * a_im;
    float t_re  = (em_re * a_re + em_im * a_im) / den; // (exp(dtA)-1)/A
    float t_im  = (em_im * a_re - em_re * a_im) / den;
    float cr = C_re[(size_t)h * 64 + n], ci = C_im[(size_t)h * 64 + n];
    cd_re[q] = cr * t_re - ci * t_im;
    cd_im[q] = cr * t_im + ci * t_re;
    s_re[q] = 0.f; s_im[q] = 0.f;
  }
  float  Dh  = Dv[h];
  float* row = xn + ((size_t)b * H + h) * L;
  for (int l0 = 0; l0 < L; l0 += 32) {
    float uv = row[l0 + lane];  // chunk read before chunk write: in-place safe
    float yv = 0.f;
    for (int j = 0; j < 32; j++) {
      float u = __shfl(uv, j, 32);
      float p = 0.f;
      #pragma unroll
      for (int q = 0; q < 2; q++) {
        float nr = w_re[q] * s_re[q] - w_im[q] * s_im[q] + u;
        float ni = w_re[q] * s_im[q] + w_im[q] * s_re[q];
        s_re[q] = nr; s_im[q] = ni;
        p += cd_re[q] * nr - cd_im[q] * ni;
      }
      p *= 2.0f;
      #pragma unroll
      for (int off = 16; off > 0; off >>= 1) p += __shfl_xor(p, off, 32);
      float y = gelu_exact(p + Dh * u);
      if (lane == j) yv = y;
    }
    row[l0 + lane] = yv;
  }
}

// ---------------------------------------------------------------------------
// 4) Fused GLU projection GEMM + residual, bf16 WMMA.
//    Z[b,o,l] = val(o)*sigmoid(gate(o+H)) + Xres[b,o,l]
//    Block tile M=128 (l), N=32 (o: value+gate pair), K step 32.
//    8 waves = 4(M) x 2(N); each wave: 2 M sub-tiles x {val,gate} = 4 WMMA.
// ---------------------------------------------------------------------------
__global__ void __launch_bounds__(256)
glu_gemm(const float* __restrict__ Y, const float* __restrict__ Xres,
         const float* __restrict__ ow, const float* __restrict__ obias,
         float* __restrict__ Z, int H, int L) {
  __shared__ __bf16 As[128][40];
  __shared__ __bf16 Bv[32][40];
  __shared__ __bf16 Bg[32][40];
  int b  = blockIdx.z;
  int l0 = blockIdx.x * 128;
  int o0 = blockIdx.y * 32;
  int tid = threadIdx.x, lane = tid & 31, wv = tid >> 5;
  int wm = (wv & 3) * 32;   // M sub-tile pair base
  int wn = (wv >> 2) * 16;  // N sub-tile
  int idx = lane & 15, hi = lane >> 4;
  v8f accv0 = {}, accv1 = {}, accg0 = {}, accg1 = {};
  const float* yb = Y + (size_t)b * H * L;

  for (int k0 = 0; k0 < H; k0 += 32) {
    // Speculative prefetch of next K-slab of the streaming A operand
    if (k0 + 32 < H)
      __builtin_prefetch(yb + (size_t)(k0 + 32 + (tid >> 3)) * L + l0 + (tid & 7) * 16, 0, 1);
    // A tile: 128(m=l) x 32(k=h); (B,H,L) contiguous along m for fixed k
    {
      int i  = tid * 16;          // 4096 elems / 256 threads
      int m  = i & 127, kk = i >> 7;
      const float4* src = (const float4*)(yb + (size_t)(k0 + kk) * L + l0 + m);
      #pragma unroll
      for (int t = 0; t < 4; t++) {
        float4 p = src[t];
        As[m + 4 * t + 0][kk] = (__bf16)p.x; As[m + 4 * t + 1][kk] = (__bf16)p.y;
        As[m + 4 * t + 2][kk] = (__bf16)p.z; As[m + 4 * t + 3][kk] = (__bf16)p.w;
      }
    }
    // W tiles: value rows [o0..o0+32), gate rows [o0+H..)
    {
      int i = tid * 4;
      int kk = i & 31, n = i >> 5;
      float4 w0 = *(const float4*)(ow + (size_t)(o0 + n) * H + k0 + kk);
      Bv[n][kk + 0] = (__bf16)w0.x; Bv[n][kk + 1] = (__bf16)w0.y;
      Bv[n][kk + 2] = (__bf16)w0.z; Bv[n][kk + 3] = (__bf16)w0.w;
      float4 w1 = *(const float4*)(ow + (size_t)(o0 + H + n) * H + k0 + kk);
      Bg[n][kk + 0] = (__bf16)w1.x; Bg[n][kk + 1] = (__bf16)w1.y;
      Bg[n][kk + 2] = (__bf16)w1.z; Bg[n][kk + 3] = (__bf16)w1.w;
    }
    __syncthreads();
    v16bf a0, a1, bv, bg;
    #pragma unroll
    for (int j = 0; j < 8; j++) {
      int kb = wmma_kbase(j, hi);
      a0[2*j] = As[wm +      idx][kb]; a0[2*j+1] = As[wm +      idx][kb + 1];
      a1[2*j] = As[wm + 16 + idx][kb]; a1[2*j+1] = As[wm + 16 + idx][kb + 1];
      bv[2*j] = Bv[wn + idx][kb];      bv[2*j+1] = Bv[wn + idx][kb + 1];
      bg[2*j] = Bg[wn + idx][kb];      bg[2*j+1] = Bg[wn + idx][kb + 1];
    }
    accv0 = __builtin_amdgcn_wmma_f32_16x16x32_bf16(false, a0, false, bv,
                                                    (short)0, accv0, false, false);
    accv1 = __builtin_amdgcn_wmma_f32_16x16x32_bf16(false, a1, false, bv,
                                                    (short)0, accv1, false, false);
    accg0 = __builtin_amdgcn_wmma_f32_16x16x32_bf16(false, a0, false, bg,
                                                    (short)0, accg0, false, false);
    accg1 = __builtin_amdgcn_wmma_f32_16x16x32_bf16(false, a1, false, bg,
                                                    (short)0, accg1, false, false);
    __syncthreads();
  }
  // Epilogue: GLU + residual. D layout: N=idx, M=j+8*hi.
  int o = o0 + wn + idx;
  float bval = obias[o], bgat = obias[o + H];
  const float* xr = Xres + ((size_t)b * H + o) * L;
  float*       zr = Z    + ((size_t)b * H + o) * L;
  #pragma unroll
  for (int j = 0; j < 8; j++) {
    int l = l0 + wm + j + 8 * hi;
    float v0 = accv0[j] + bval, g0 = accg0[j] + bgat;
    zr[l] = v0 * sigmoidf(g0) + xr[l];
    float v1 = accv1[j] + bval, g1 = accg1[j] + bgat;
    zr[l + 16] = v1 * sigmoidf(g1) + xr[l + 16];
  }
}

// ---------------------------------------------------------------------------
// 5) Fused AvgPool1d(2) + linear projection GEMM + GELU, bf16 WMMA.
//    Xout[b,o,lp] = gelu( sum_k lw[o,k] * 0.5*(Z[b,k,2lp]+Z[b,k,2lp+1]) + lb[o] )
//    Block tile M=128 (lp), N=32, K step 32; each wave 2 M sub-tiles = 2 WMMA.
// ---------------------------------------------------------------------------
__global__ void __launch_bounds__(256)
pool_gemm_gelu(const float* __restrict__ Zin, const float* __restrict__ lw,
               const float* __restrict__ lbias, float* __restrict__ Xout,
               int H, int L, int Ho, int Lo) {
  __shared__ __bf16 As[128][40];
  __shared__ __bf16 Bs[32][40];
  int b   = blockIdx.z;
  int lp0 = blockIdx.x * 128;
  int o0  = blockIdx.y * 32;
  int tid = threadIdx.x, lane = tid & 31, wv = tid >> 5;
  int wm = (wv & 3) * 32;
  int wn = (wv >> 2) * 16;
  int idx = lane & 15, hi = lane >> 4;
  v8f acc0 = {}, acc1 = {};
  const float* zb = Zin + (size_t)b * H * L;

  for (int k0 = 0; k0 < H; k0 += 32) {
    if (k0 + 32 < H)
      __builtin_prefetch(zb + (size_t)(k0 + 32 + (tid >> 3)) * L + 2 * lp0 + (tid & 7) * 32, 0, 1);
    {
      int i = tid * 16;
      int m = i & 127, kk = i >> 7;
      const float2* src = (const float2*)(zb + (size_t)(k0 + kk) * L + 2 * (lp0 + m));
      #pragma unroll
      for (int t = 0; t < 16; t++) {
        float2 p = src[t];
        As[m + t][kk] = (__bf16)(0.5f * (p.x + p.y));  // fused AvgPool(2)
      }
    }
    {
      int i = tid * 4;
      int kk = i & 31, n = i >> 5;
      float4 w0 = *(const float4*)(lw + (size_t)(o0 + n) * H + k0 + kk);
      Bs[n][kk + 0] = (__bf16)w0.x; Bs[n][kk + 1] = (__bf16)w0.y;
      Bs[n][kk + 2] = (__bf16)w0.z; Bs[n][kk + 3] = (__bf16)w0.w;
    }
    __syncthreads();
    v16bf a0, a1, bb;
    #pragma unroll
    for (int j = 0; j < 8; j++) {
      int kb = wmma_kbase(j, hi);
      a0[2*j] = As[wm +      idx][kb]; a0[2*j+1] = As[wm +      idx][kb + 1];
      a1[2*j] = As[wm + 16 + idx][kb]; a1[2*j+1] = As[wm + 16 + idx][kb + 1];
      bb[2*j] = Bs[wn + idx][kb];      bb[2*j+1] = Bs[wn + idx][kb + 1];
    }
    acc0 = __builtin_amdgcn_wmma_f32_16x16x32_bf16(false, a0, false, bb,
                                                   (short)0, acc0, false, false);
    acc1 = __builtin_amdgcn_wmma_f32_16x16x32_bf16(false, a1, false, bb,
                                                   (short)0, acc1, false, false);
    __syncthreads();
  }
  int o = o0 + wn + idx;
  float bo = lbias[o];
  float* xr = Xout + ((size_t)b * Ho + o) * Lo;
  #pragma unroll
  for (int j = 0; j < 8; j++) {
    int lp = lp0 + wm + j + 8 * hi;
    xr[lp]      = gelu_exact(acc0[j] + bo);
    xr[lp + 16] = gelu_exact(acc1[j] + bo);
  }
}

// ---------------------------------------------------------------------------
// 6) Mean over L: one wave per (b,h)
// ---------------------------------------------------------------------------
__global__ void __launch_bounds__(256)
mean_l(const float* __restrict__ X, float* __restrict__ out, int H, int L) {
  int lane = threadIdx.x & 31;
  int wid  = blockIdx.x * (blockDim.x >> 5) + (threadIdx.x >> 5);
  int b = wid / H, h = wid % H;
  const float* row = X + ((size_t)b * H + h) * L;
  float s = 0.f;
  for (int l = lane; l < L; l += 32) s += row[l];
  #pragma unroll
  for (int off = 16; off > 0; off >>= 1) s += __shfl_xor(s, off, 32);
  if (lane == 0) out[(size_t)b * H + h] = s / (float)L;
}

// ---------------------------------------------------------------------------
// 7) Decoder: out[b,o] = xm[b,:] . dec_w[o,:] + dec_b[o]   (16 x 1000 x 128)
// ---------------------------------------------------------------------------
__global__ void decoder_gemv(const float* __restrict__ xm,
                             const float* __restrict__ w,
                             const float* __restrict__ bias,
                             float* __restrict__ out, int H, int O, int Bn) {
  int i = blockIdx.x * blockDim.x + threadIdx.x;
  if (i >= Bn * O) return;
  int b = i / O, o = i % O;
  float s = bias[o];
  for (int k = 0; k < H; k++) s += xm[(size_t)b * H + k] * w[(size_t)o * H + k];
  out[i] = s;
}

// ---------------------------------------------------------------------------
// Host driver
// ---------------------------------------------------------------------------
extern "C" void kernel_launch(void* const* d_in, const int* in_sizes, int n_in,
                              void* d_out, int out_size, void* d_ws, size_t ws_size,
                              hipStream_t stream) {
  (void)in_sizes; (void)n_in; (void)out_size; (void)ws_size;
  const int Bn = 16, NLAY = 3, DOUT = 1000;
  const int Hs[3] = {1024, 512, 256};
  const int Ls[3] = {2048, 1024, 512};

  const float* x1 = (const float*)d_in[0];

  // Workspace: three rotating (B,H,L) f32 buffers sized for layer 0 + mean buf
  const size_t BIG = (size_t)Bn * 1024 * 2048;  // elements
  float* buf0 = (float*)d_ws;                   // x (residual source / next x)
  float* buf1 = buf0 + BIG;                     // LN out -> S4 out (in place)
  float* buf2 = buf1 + BIG;                     // GLU+residual out
  float* xm   = buf2 + BIG;                     // (B, 128) mean

  // x1 (B,L,1024) -> buf0 (B,1024,L)
  {
    dim3 grid(Hs[0] / 32, Ls[0] / 32, Bn), block(32, 8);
    transpose_blh_to_bhl<<<grid, block, 0, stream>>>(x1, buf0, Hs[0], Ls[0]);
  }

  for (int lay = 0; lay < NLAY; lay++) {
    const int base = 2 + 10 * lay;
    const float* ln_g   = (const float*)d_in[base + 0];
    const float* ln_b   = (const float*)d_in[base + 1];
    const float* log_dt = (const float*)d_in[base + 2];
    const float* C_re   = (const float*)d_in[base + 3];
    const float* C_im   = (const float*)d_in[base + 4];
    const float* Dv     = (const float*)d_in[base + 5];
    const float* ow     = (const float*)d_in[base + 6];
    const float* ob     = (const float*)d_in[base + 7];
    const float* lw     = (const float*)d_in[base + 8];
    const float* lb     = (const float*)d_in[base + 9];
    const int H = Hs[lay], L = Ls[lay];
    const int Ho = H / 2, Lo = L / 2;

    layernorm_bhl<<<dim3(L / 32, Bn), 256, 0, stream>>>(buf0, ln_g, ln_b, buf1, H, L);
    s4_scan<<<(Bn * H) / 8, 256, 0, stream>>>(buf1, log_dt, C_re, C_im, Dv, H, L);
    glu_gemm<<<dim3(L / 128, H / 32, Bn), 256, 0, stream>>>(buf1, buf0, ow, ob, buf2, H, L);
    pool_gemm_gelu<<<dim3(Lo / 128, Ho / 32, Bn), 256, 0, stream>>>(buf2, lw, lb, buf0,
                                                                    H, L, Ho, Lo);
  }

  // buf0 now (B,128,256)
  mean_l<<<(Bn * Hs[2] / 2) / 8, 256, 0, stream>>>(buf0, xm, 128, 256);
  const float* dec_w = (const float*)d_in[32];
  const float* dec_b = (const float*)d_in[33];
  decoder_gemv<<<(Bn * DOUT + 255) / 256, 256, 0, stream>>>(xm, dec_w, dec_b,
                                                            (float*)d_out, 128, DOUT, Bn);
}
